// WaveformSpatializer_33423435497712
// MI455X (gfx1250) — matrix operand order repaired and checked by reading, not compiled, not run on hardware
//
#include <hip/hip_runtime.h>
#include <hip/hip_bf16.h>

typedef __attribute__((ext_vector_type(16))) _Float16 v16h;
typedef __attribute__((ext_vector_type(8)))  _Float16 v8h;
typedef __attribute__((ext_vector_type(8)))  float    v8f;

#define B_SZ  32
#define T_SZ  262144
#define K_SZ  100
#define VDIM  7
#define VHID  128
#define VOUT  256
#define FLEN  512
#define FHID  512

#define STRIP 8192
#define HALO  256
#define MONO_LDS (STRIP + 2*HALO)        // 8704 f16 samples (raw count)

// LDS bank-conflict padding: +8 halfs (16B) per 128-half (256B) row.
#define PIDX(i) ((i) + (((i) >> 7) << 3))
#define MONO_LDS_BYTES ((MONO_LDS/128) * 272)          // 68 rows * 272B = 18496

// Banded-B chunk: 32 lanes x 32B, +16B pad per 8 lanes -> 1088 B / chunk.
#define BCHUNK_HALFS 544
#define BCH_HALFS    (32*BCHUNK_HALFS)                 // 17408 halfs / channel
#define SB_BYTES     (2*BCH_HALFS*2)                   // 69632 B (both channels)

struct P {
    const float* view;
    const float* w[3];  const float* b[3];
    const float* fw[2][3]; const float* fb[2][3];
    _Float16* Bws;
};

// --------------------------- CDNA5 async copy helpers -----------------------
__device__ __forceinline__ void async_ld_b128(unsigned lds_off, const void* g) {
    asm volatile("global_load_async_to_lds_b128 %0, %1, off"
                 :: "v"(lds_off), "v"(g) : "memory");
}
__device__ __forceinline__ void wait_async0() {
#if __has_builtin(__builtin_amdgcn_s_wait_asynccnt)
    __builtin_amdgcn_s_wait_asynccnt(0);
#else
    asm volatile("s_wait_asynccnt 0x0" ::: "memory");
#endif
}

// ---------------------------------------------------------------------------
// Kernel 1: view MLP + filter generators + L1 norm + padded banded-B tiles.
// Banded B: B_c[kk,n] = f[16c+kk-n] if 0 <= kk-n < 16 else 0, in WMMA
// B-operand lane order (lane half h holds k = 8h..8h+7, then 16+8h..16+8h+7), 16B pad / 8 lanes.
// ---------------------------------------------------------------------------
__global__ __launch_bounds__(512) void k_filters(P p) {
    __shared__ float sVm[VDIM], sH1[VHID], sH2[VHID], sFeat[VOUT];
    __shared__ float sG1[FHID], sG2[FHID], sF[FLEN], sRed[512];
    __shared__ float sInv;
    const int b = blockIdx.x, tid = threadIdx.x;

    if (tid < VDIM) {
        float s = 0.f;
        const float* v = p.view + (size_t)(b*VDIM + tid)*K_SZ;
        for (int k = 0; k < K_SZ; ++k) s += v[k];
        sVm[tid] = s * (1.0f / (float)K_SZ);
    }
    __syncthreads();
    if (tid < VHID) {
        float s = p.b[0][tid];
        for (int j = 0; j < VDIM; ++j) s += p.w[0][tid*VDIM + j] * sVm[j];
        sH1[tid] = fmaxf(s, 0.f);
    }
    __syncthreads();
    if (tid < VHID) {
        float s = p.b[1][tid];
        for (int j = 0; j < VHID; ++j) s += p.w[1][tid*VHID + j] * sH1[j];
        sH2[tid] = fmaxf(s, 0.f);
    }
    __syncthreads();
    if (tid < VOUT) {
        float s = p.b[2][tid];
        for (int j = 0; j < VHID; ++j) s += p.w[2][tid*VHID + j] * sH2[j];
        sFeat[tid] = s;
    }
    __syncthreads();

    for (int ch = 0; ch < 2; ++ch) {
        float s = p.fb[ch][0][tid];
        for (int j = 0; j < VOUT; ++j) s += p.fw[ch][0][tid*VOUT + j] * sFeat[j];
        sG1[tid] = fmaxf(s, 0.f);
        __syncthreads();
        s = p.fb[ch][1][tid];
        for (int j = 0; j < FHID; ++j) s += p.fw[ch][1][tid*FHID + j] * sG1[j];
        sG2[tid] = fmaxf(s, 0.f);
        __syncthreads();
        s = p.fb[ch][2][tid];
        for (int j = 0; j < FHID; ++j) s += p.fw[ch][2][tid*FHID + j] * sG2[j];
        float f = tanhf(s);
        sF[tid]   = f;
        sRed[tid] = fabsf(f);
        __syncthreads();
        for (int off = 256; off > 0; off >>= 1) {
            if (tid < off) sRed[tid] += sRed[tid + off];
            __syncthreads();
        }
        if (tid == 0) sInv = 1.0f / (sRed[0] + 1e-8f);
        __syncthreads();
        const float inv = sInv;

        _Float16* dst = p.Bws + (size_t)(b*2 + ch) * BCH_HALFS;
        for (int e = tid; e < BCH_HALFS; e += 512) {       // 17408 % 512 == 0
            int c  = e / BCHUNK_HALFS;
            int r  = e - c*BCHUNK_HALFS;
            int g  = r / 136;                              // 8 lanes + pad = 136 halfs
            int rg = r - g*136;
            float v = 0.f;
            if (rg < 128) {
                int lane = g*8 + (rg >> 4), j = rg & 15;
                int kk = ((lane >> 4) << 3) + j + ((j >> 3) << 3);
                int n  = lane & 15;
                int d  = kk - n;
                if (d >= 0 && d < 16) v = sF[(c << 4) + d] * inv;
            }
            dst[e] = (_Float16)v;
        }
        __syncthreads();
    }
}

// ---------------------------------------------------------------------------
// Kernel 2: mono f32 -> f16.
// ---------------------------------------------------------------------------
struct H4 { _Float16 a, b, c, d; };
__global__ __launch_bounds__(256) void k_tohalf(const float* __restrict__ x,
                                                _Float16* __restrict__ y, int n4) {
    int i = blockIdx.x * 256 + threadIdx.x;
    if (i < n4) {
        float4 v = ((const float4*)x)[i];
        H4 o; o.a = (_Float16)v.x; o.b = (_Float16)v.y;
        o.c = (_Float16)v.z; o.d = (_Float16)v.w;
        ((H4*)y)[i] = o;
    }
}

// ---------------------------------------------------------------------------
// Kernel 3: FIR correlation as banded-Toeplitz WMMA GEMM.
// grid = 32 batches x 32 strips; block = 256 threads (8 waves).
// Async staging; per wave: 4 time tiles x 2 channels, 32 tap-chunks,
// ping-pong A-operand halves (hi(c) == lo(c+1)) -> 8 ds_load_b128 per 8 WMMA.
// ---------------------------------------------------------------------------
__global__ __launch_bounds__(256) void k_conv(const _Float16* __restrict__ monoH,
                                              const _Float16* __restrict__ Bws,
                                              float* __restrict__ out) {
    extern __shared__ char smemRaw[];
    _Float16* sB_h    = (_Float16*)smemRaw;                    // SB_BYTES
    _Float16* sMono_h = (_Float16*)(smemRaw + SB_BYTES);       // MONO_LDS_BYTES

    const int b   = blockIdx.x >> 5;
    const int t0s = (blockIdx.x & 31) * STRIP;
    const int tid = threadIdx.x;

    const unsigned ldsB    = (unsigned)(unsigned long long)(void*)sB_h;
    const unsigned ldsMono = (unsigned)(unsigned long long)(void*)sMono_h;

    // --- Stage banded-B (both channels, already padded in ws): linear async copy.
    {
        const _Float16* src = Bws + (size_t)(b*2) * BCH_HALFS;
        for (int u = tid; u < SB_BYTES/16; u += 256)           // 4352 units
            async_ld_b128(ldsB + (unsigned)u*16u, (const void*)(src + u*8));
    }
    // --- Stage mono strip + halo (padded rows: +16B per 16 units).
    const _Float16* mrow = monoH + (size_t)b * T_SZ;
    if (t0s >= HALO && t0s + STRIP + HALO <= T_SZ) {
        const _Float16* src = mrow + t0s - HALO;
        for (int u = tid; u < (MONO_LDS*2)/16; u += 256) {     // 1088 units
            unsigned dst = ldsMono + (unsigned)(u*16 + (u >> 4)*16);
            async_ld_b128(dst, (const void*)(src + u*8));
        }
    } else {
        for (int i = tid; i < MONO_LDS; i += 256) {
            int pp = t0s - HALO + i;
            sMono_h[PIDX(i)] = (pp >= 0 && pp < T_SZ) ? mrow[pp] : (_Float16)0.f;
        }
    }
    wait_async0();
    __syncthreads();

    const int wv = tid >> 5, lane = tid & 31;
    const int m  = lane & 15, h = lane >> 4;
    const int abase = 16*m + 8*h;
    const int boff0 = (lane << 4) + ((lane >> 3) << 3);        // padded B lane offset

    v8f acc[2][4] = {};
    v8h curA[4], curB[4];
    #pragma unroll
    for (int i = 0; i < 4; ++i)
        curA[i] = *(const v8h*)(sMono_h + PIDX((((wv << 2) + i) << 8) + abase));

    for (int c = 0; c < 32; c += 2) {
        // ---------------- even chunk c: A = [curA, curB] ----------------
        {
            const int bo = c*BCHUNK_HALFS + boff0;
            v8h bl0 = *(const v8h*)(sB_h + bo);
            v8h bl1 = *(const v8h*)(sB_h + bo + 8);
            v8h br0 = *(const v8h*)(sB_h + BCH_HALFS + bo);
            v8h br1 = *(const v8h*)(sB_h + BCH_HALFS + bo + 8);
            v16h bl = __builtin_shufflevector(bl0, bl1, 0,1,2,3,4,5,6,7,8,9,10,11,12,13,14,15);
            v16h br = __builtin_shufflevector(br0, br1, 0,1,2,3,4,5,6,7,8,9,10,11,12,13,14,15);
            v16h a[4];
            #pragma unroll
            for (int i = 0; i < 4; ++i) {
                int idxn = (((wv << 2) + i) << 8) + abase + ((c + 1) << 4);
                curB[i] = *(const v8h*)(sMono_h + PIDX(idxn));
                a[i] = __builtin_shufflevector(curA[i], curB[i],
                                               0,1,2,3,4,5,6,7,8,9,10,11,12,13,14,15);
            }
            #pragma unroll
            for (int i = 0; i < 4; ++i) {
                acc[0][i] = __builtin_amdgcn_wmma_f32_16x16x32_f16(
                    false, a[i], false, bl, (short)0, acc[0][i], false, false);
                acc[1][i] = __builtin_amdgcn_wmma_f32_16x16x32_f16(
                    false, a[i], false, br, (short)0, acc[1][i], false, false);
            }
        }
        // ---------------- odd chunk c+1: A = [curB, curA'] ----------------
        {
            const int bo = (c + 1)*BCHUNK_HALFS + boff0;
            v8h bl0 = *(const v8h*)(sB_h + bo);
            v8h bl1 = *(const v8h*)(sB_h + bo + 8);
            v8h br0 = *(const v8h*)(sB_h + BCH_HALFS + bo);
            v8h br1 = *(const v8h*)(sB_h + BCH_HALFS + bo + 8);
            v16h bl = __builtin_shufflevector(bl0, bl1, 0,1,2,3,4,5,6,7,8,9,10,11,12,13,14,15);
            v16h br = __builtin_shufflevector(br0, br1, 0,1,2,3,4,5,6,7,8,9,10,11,12,13,14,15);
            v16h a[4];
            #pragma unroll
            for (int i = 0; i < 4; ++i) {
                int idxn = (((wv << 2) + i) << 8) + abase + ((c + 2) << 4);
                curA[i] = *(const v8h*)(sMono_h + PIDX(idxn));
                a[i] = __builtin_shufflevector(curB[i], curA[i],
                                               0,1,2,3,4,5,6,7,8,9,10,11,12,13,14,15);
            }
            #pragma unroll
            for (int i = 0; i < 4; ++i) {
                acc[0][i] = __builtin_amdgcn_wmma_f32_16x16x32_f16(
                    false, a[i], false, bl, (short)0, acc[0][i], false, false);
                acc[1][i] = __builtin_amdgcn_wmma_f32_16x16x32_f16(
                    false, a[i], false, br, (short)0, acc[1][i], false, false);
            }
        }
    }

    // D layout: lane holds n = lane&15, rows m = r + 8*(lane>>4).
    #pragma unroll
    for (int ch = 0; ch < 2; ++ch) {
        #pragma unroll
        for (int i = 0; i < 4; ++i) {
            float* ob = out + ((size_t)(b*2 + ch)) * T_SZ + t0s
                        + (((wv << 2) + i) << 8) + (h << 7) + m;
            #pragma unroll
            for (int r = 0; r < 8; ++r) ob[r << 4] = acc[ch][i][r];
        }
    }
}

// ---------------------------------------------------------------------------
extern "C" void kernel_launch(void* const* d_in, const int* in_sizes, int n_in,
                              void* d_out, int out_size, void* d_ws, size_t ws_size,
                              hipStream_t stream) {
    (void)in_sizes; (void)n_in; (void)out_size; (void)ws_size;
    const float* mono = (const float*)d_in[0];

    P p;
    p.view = (const float*)d_in[1];
    p.w[0] = (const float*)d_in[2]; p.b[0] = (const float*)d_in[3];
    p.w[1] = (const float*)d_in[4]; p.b[1] = (const float*)d_in[5];
    p.w[2] = (const float*)d_in[6]; p.b[2] = (const float*)d_in[7];
    for (int ch = 0; ch < 2; ++ch)
        for (int l = 0; l < 3; ++l) {
            p.fw[ch][l] = (const float*)d_in[8 + ch*6 + l*2];
            p.fb[ch][l] = (const float*)d_in[9 + ch*6 + l*2];
        }

    _Float16* Bws   = (_Float16*)d_ws;                         // ~2.2 MiB padded
    _Float16* monoH = (_Float16*)((char*)d_ws + ((size_t)4 << 20));
    p.Bws = Bws;

    k_filters<<<B_SZ, 512, 0, stream>>>(p);

    int n4 = (B_SZ * T_SZ) / 4;
    k_tohalf<<<(n4 + 255)/256, 256, 0, stream>>>(mono, monoH, n4);

    size_t smem = (size_t)SB_BYTES + (size_t)MONO_LDS_BYTES;   // 88128 B
    k_conv<<<B_SZ*32, 256, smem, stream>>>(monoH, Bws, (float*)d_out);
}
